// GatedLinearAttention_41068477285024
// MI455X (gfx1250) — matrix operand order repaired and checked by reading, not compiled, not run on hardware
//
#include <hip/hip_runtime.h>
#include <hip/hip_bf16.h>

// ---------------------------------------------------------------------------
// Gated Linear Attention for MI455X (gfx1250, wave32, WMMA bf16 16x16x32)
// GEMM path: async global->LDS double buffering (ASYNCcnt) + WMMA from LDS.
// ---------------------------------------------------------------------------
#define GLA_B   2
#define GLA_T   2048
#define GLA_D   1024
#define GLA_H   16
#define GLA_HD  64
#define GLA_M   (GLA_B * GLA_T)   // 4096

#define BK      32                // K slab per pipeline stage
#define ROWP    40                // padded LDS row stride in ushorts (80 B)

typedef __attribute__((ext_vector_type(16))) __bf16 v16bf;
typedef __attribute__((ext_vector_type(8)))  __bf16 v8bf;
typedef __attribute__((ext_vector_type(8)))  float  v8f;
typedef __attribute__((ext_vector_type(4)))  int    v4i;

typedef __attribute__((address_space(1))) v4i gv4i;   // global int4
typedef __attribute__((address_space(3))) v4i lv4i;   // LDS int4

union ABFrag { v16bf v; v8bf h[2]; };

__device__ __forceinline__ unsigned short f2bf(float f) {
    unsigned u = __float_as_uint(f);
    unsigned r = (u + 0x7FFFu + ((u >> 16) & 1u)) >> 16;   // RNE
    return (unsigned short)r;
}
__device__ __forceinline__ float bf2f(unsigned short b) {
    return __uint_as_float(((unsigned)b) << 16);
}

// --- async 16B global -> LDS copy (CDNA5 GLOBAL_LOAD_ASYNC_TO_LDS_B128) ----
#if __has_builtin(__builtin_amdgcn_global_load_async_to_lds_b128)
#define GLA_ASYNC_CP 1
#else
#define GLA_ASYNC_CP 0
#endif

__device__ __forceinline__ void cp16(const unsigned short* g, unsigned short* l) {
#if GLA_ASYNC_CP
    __builtin_amdgcn_global_load_async_to_lds_b128((gv4i*)g, (lv4i*)l, 0, 0);
#else
    *(v8bf*)l = *(const v8bf*)g;
#endif
}

__device__ __forceinline__ void async_wait_all() {
#if GLA_ASYNC_CP
#if __has_builtin(__builtin_amdgcn_s_wait_asynccnt)
    __builtin_amdgcn_s_wait_asynccnt(0);
#else
    asm volatile("s_wait_asynccnt 0x0" ::: "memory");
#endif
#endif
}

// ---------------------------------------------------------------------------
// fp32 -> bf16 conversion
// ---------------------------------------------------------------------------
__global__ void cvt_f32_bf16(const float* __restrict__ src,
                             unsigned short* __restrict__ dst, int n) {
    int i = blockIdx.x * blockDim.x + threadIdx.x;
    if (i < n) dst[i] = f2bf(src[i]);
}

// ---------------------------------------------------------------------------
// bf16 WMMA GEMM:  C[m][n] = sum_k A[m][k] * W[n][k]   (W is (N,K) row-major)
// Block: 256 threads = 8 waves; block tile 128(M) x 128(N); wave tile 32x64.
// LDS double buffer: 2 x (128 x 32) bf16 slabs for A and B (padded rows).
// MODE 0: store fp32.  MODE 1: store bf16, sigmoid applied for n >= gstart.
// ---------------------------------------------------------------------------
template <int MODE>
__global__ __launch_bounds__(256)
void gemm_bf16_wmma(const unsigned short* __restrict__ A,
                    const unsigned short* __restrict__ W,
                    void* __restrict__ Cout,
                    int M, int N, int K, int gstart) {
    __shared__ unsigned short lA[2][128 * ROWP];
    __shared__ unsigned short lB[2][128 * ROWP];

    const int tid  = threadIdx.x;
    const int lane = tid & 31;
    const int wave = tid >> 5;

    const int m0 = blockIdx.x * 128 + (wave & 3) * 32;   // wave M origin (global)
    const int n0 = blockIdx.y * 128 + (wave >> 2) * 64;  // wave N origin (global)
    const int wmL = (wave & 3) * 32;                     // wave M origin (block-local)
    const int wnL = (wave >> 2) * 64;                    // wave N origin (block-local)

    // ISA 7.12.2 wave32 operand layouts:
    // A 16x32 bf16: lane L holds row M=L%16; K in {kA..kA+7, kA+16..kA+23}, kA=(L<16?0:8)
// B 32x16 bf16: lane L holds col N=L%16; K in {kB..kB+7, kB+16..kB+23}, kB=(L<16?0:8)
    const int mrow  = lane & 15;
    const int koffA = (lane >> 4) * 8;
    const int koffB = (lane >> 4) * 8;

    // copy descriptors: each thread moves two 16B chunks of A and two of B
    const int cA0 = tid, cA1 = tid + 256;                // chunk = (row, 16B segment)
    const unsigned short* gA = A + (size_t)(blockIdx.x * 128) * K;
    const unsigned short* gB = W + (size_t)(blockIdx.y * 128) * K;

    auto issue = [&](int buf, int k) {
        cp16(gA + (size_t)(cA0 >> 2) * K + k + (cA0 & 3) * 8,
             &lA[buf][(cA0 >> 2) * ROWP + (cA0 & 3) * 8]);
        cp16(gA + (size_t)(cA1 >> 2) * K + k + (cA1 & 3) * 8,
             &lA[buf][(cA1 >> 2) * ROWP + (cA1 & 3) * 8]);
        cp16(gB + (size_t)(cA0 >> 2) * K + k + (cA0 & 3) * 8,
             &lB[buf][(cA0 >> 2) * ROWP + (cA0 & 3) * 8]);
        cp16(gB + (size_t)(cA1 >> 2) * K + k + (cA1 & 3) * 8,
             &lB[buf][(cA1 >> 2) * ROWP + (cA1 & 3) * 8]);
    };

    v8f acc[2][4];
#pragma unroll
    for (int i = 0; i < 2; ++i)
#pragma unroll
        for (int j = 0; j < 4; ++j) acc[i][j] = v8f{};

    const int nk = K / BK;
    issue(0, 0);

    for (int kt = 0; kt < nk; ++kt) {
        const int cur = kt & 1;
        async_wait_all();          // this wave's copies into buf `cur` landed
        __syncthreads();           // all waves' copies landed

        if (kt + 1 < nk) issue(cur ^ 1, (kt + 1) * BK);  // overlap next slab

        const unsigned short* bufA = lA[cur];
        const unsigned short* bufB = lB[cur];

        ABFrag a0, a1, b[4];
        const int r0 = (wmL + mrow) * ROWP;
        const int r1 = r0 + 16 * ROWP;
        a0.h[0] = *(const v8bf*)(bufA + r0 + koffA);
        a0.h[1] = *(const v8bf*)(bufA + r0 + koffA + 16);
        a1.h[0] = *(const v8bf*)(bufA + r1 + koffA);
        a1.h[1] = *(const v8bf*)(bufA + r1 + koffA + 16);
#pragma unroll
        for (int j = 0; j < 4; ++j) {
            const int nr = (wnL + j * 16 + mrow) * ROWP;
            b[j].h[0] = *(const v8bf*)(bufB + nr + koffB);
            b[j].h[1] = *(const v8bf*)(bufB + nr + koffB + 16);
        }

#pragma unroll
        for (int j = 0; j < 4; ++j) {
            acc[0][j] = __builtin_amdgcn_wmma_f32_16x16x32_bf16(
                false, a0.v, false, b[j].v, (short)0, acc[0][j], false, false);
            acc[1][j] = __builtin_amdgcn_wmma_f32_16x16x32_bf16(
                false, a1.v, false, b[j].v, (short)0, acc[1][j], false, false);
        }
        __syncthreads();           // everyone done reading buf `cur`
    }

    // C/D layout: VGPR r -> M = r + 8*(lane/16); N = lane%16
    const int msub = (lane >> 4) * 8;
    const int ncol = lane & 15;
#pragma unroll
    for (int i = 0; i < 2; ++i) {
#pragma unroll
        for (int j = 0; j < 4; ++j) {
            const int n = n0 + j * 16 + ncol;
#pragma unroll
            for (int r = 0; r < 8; ++r) {
                const int m = m0 + i * 16 + msub + r;
                float v = acc[i][j][r];
                if (MODE == 0) {
                    ((float*)Cout)[(size_t)m * N + n] = v;
                } else {
                    if (n >= gstart) v = 1.0f / (1.0f + __expf(-v));  // gate
                    ((unsigned short*)Cout)[(size_t)m * N + n] = f2bf(v);
                }
            }
        }
    }
}

// ---------------------------------------------------------------------------
// Recurrent scan: one block per (b, head), 64 threads, thread e owns h[:, e].
// QKVA layout: (M, 4*D) bf16 rows; cols [0,D)=q, [D,2D)=k, [2D,3D)=v, [3D,4D)=a.
// ---------------------------------------------------------------------------
__global__ __launch_bounds__(64)
void gla_scan(const unsigned short* __restrict__ QKVA, float* __restrict__ Y) {
    const int bh = blockIdx.x;
    const int b  = bh >> 4;
    const int h  = bh & 15;
    const int e  = threadIdx.x;            // 0..63 : state column

    __shared__ float sq[GLA_HD], sk[GLA_HD], sa[GLA_HD];

    float hst[GLA_HD];                     // h[d][e] for all d, in registers
#pragma unroll
    for (int d = 0; d < GLA_HD; ++d) hst[d] = 0.0f;

    const int ld = 4 * GLA_D;
    const unsigned short* base = QKVA + (size_t)b * GLA_T * ld + h * GLA_HD;
    float* yout = Y + (size_t)b * GLA_T * GLA_D + h * GLA_HD + e;

    for (int t = 0; t < GLA_T; ++t) {
        const unsigned short* row = base + (size_t)t * ld;
        const float vv = bf2f(row[2 * GLA_D + e]);
        sq[e] = bf2f(row[e]);
        sk[e] = bf2f(row[GLA_D + e]);
        sa[e] = bf2f(row[3 * GLA_D + e]);
        __syncthreads();

        float y = 0.0f;
#pragma unroll
        for (int d = 0; d < GLA_HD; ++d) {
            hst[d] = fmaf(sk[d], vv, sa[d] * hst[d]);   // h = diag(a)h + k v^T
            y      = fmaf(sq[d], hst[d], y);            // y = q^T h
        }
        yout[(size_t)t * GLA_D] = y;
        __syncthreads();
    }
}

// ---------------------------------------------------------------------------
// LayerNorm over D=1024 per row, emit bf16 (A operand of the final GEMM).
// ---------------------------------------------------------------------------
__global__ __launch_bounds__(256)
void layernorm_bf16(const float* __restrict__ Y,
                    const float* __restrict__ gamma,
                    const float* __restrict__ beta,
                    unsigned short* __restrict__ Out) {
    const int row = blockIdx.x;
    const int tid = threadIdx.x;
    const float* y = Y + (size_t)row * GLA_D;

    float vals[4], s = 0.0f, s2 = 0.0f;
#pragma unroll
    for (int i = 0; i < 4; ++i) {
        float v = y[tid + i * 256];
        vals[i] = v;
        s  += v;
        s2 += v * v;
    }

    __shared__ float red[256];
    red[tid] = s;
    __syncthreads();
    for (int off = 128; off > 0; off >>= 1) {
        if (tid < off) red[tid] += red[tid + off];
        __syncthreads();
    }
    const float mean = red[0] * (1.0f / GLA_D);
    __syncthreads();
    red[tid] = s2;
    __syncthreads();
    for (int off = 128; off > 0; off >>= 1) {
        if (tid < off) red[tid] += red[tid + off];
        __syncthreads();
    }
    const float var = red[0] * (1.0f / GLA_D) - mean * mean;
    const float inv = rsqrtf(var + 1e-5f);

#pragma unroll
    for (int i = 0; i < 4; ++i) {
        const int col = tid + i * 256;
        const float o = (vals[i] - mean) * inv * gamma[col] + beta[col];
        Out[(size_t)row * GLA_D + col] = f2bf(o);
    }
}

// ---------------------------------------------------------------------------
// Host-side launch
// ---------------------------------------------------------------------------
extern "C" void kernel_launch(void* const* d_in, const int* in_sizes, int n_in,
                              void* d_out, int out_size, void* d_ws, size_t ws_size,
                              hipStream_t stream) {
    (void)in_sizes; (void)n_in; (void)out_size; (void)ws_size;

    const float* x     = (const float*)d_in[0];
    const float* Wq    = (const float*)d_in[1];
    const float* Wk    = (const float*)d_in[2];
    const float* Wv    = (const float*)d_in[3];
    const float* Wg    = (const float*)d_in[4];
    const float* Wo    = (const float*)d_in[5];
    const float* gamma = (const float*)d_in[6];
    const float* beta  = (const float*)d_in[7];

    const int M = GLA_M, D = GLA_D;
    const size_t DD = (size_t)D * D;

    // Workspace carve-up (total ~74 MB)
    char* p = (char*)d_ws;
    unsigned short* xb   = (unsigned short*)p;  p += (size_t)M * D * 2;      //  8 MB
    unsigned short* Wcat = (unsigned short*)p;  p += 4 * DD * 2;             //  8 MB
    unsigned short* Wob  = (unsigned short*)p;  p += DD * 2;                 //  2 MB
    unsigned short* QKVA = (unsigned short*)p;  p += (size_t)M * 4 * D * 2;  // 32 MB
    float*          Yrec = (float*)p;           p += (size_t)M * D * 4;      // 16 MB
    unsigned short* Yln  = (unsigned short*)p;  p += (size_t)M * D * 2;      //  8 MB

    // 1) fp32 -> bf16 conversions (weights concatenated [Wq;Wk;Wv;Wg])
    const int cb = 256;
    cvt_f32_bf16<<<(M * D + cb - 1) / cb, cb, 0, stream>>>(x, xb, M * D);
    cvt_f32_bf16<<<((int)DD + cb - 1) / cb, cb, 0, stream>>>(Wq, Wcat + 0 * DD, (int)DD);
    cvt_f32_bf16<<<((int)DD + cb - 1) / cb, cb, 0, stream>>>(Wk, Wcat + 1 * DD, (int)DD);
    cvt_f32_bf16<<<((int)DD + cb - 1) / cb, cb, 0, stream>>>(Wv, Wcat + 2 * DD, (int)DD);
    cvt_f32_bf16<<<((int)DD + cb - 1) / cb, cb, 0, stream>>>(Wg, Wcat + 3 * DD, (int)DD);
    cvt_f32_bf16<<<((int)DD + cb - 1) / cb, cb, 0, stream>>>(Wo, Wob, (int)DD);

    // 2) Fused QKVG GEMM (N = 4096), sigmoid on gate quarter, bf16 out
    gemm_bf16_wmma<1><<<dim3(M / 128, (4 * D) / 128), 256, 0, stream>>>(
        xb, Wcat, QKVA, M, 4 * D, D, /*gstart=*/3 * D);

    // 3) Recurrent gated scan: one block per (b, head)
    gla_scan<<<GLA_B * GLA_H, GLA_HD, 0, stream>>>(QKVA, Yrec);

    // 4) LayerNorm -> bf16
    layernorm_bf16<<<M, 256, 0, stream>>>(Yrec, gamma, beta, Yln);

    // 5) Output GEMM -> fp32 d_out
    gemm_bf16_wmma<0><<<dim3(M / 128, D / 128), 256, 0, stream>>>(
        Yln, Wob, (float*)d_out, M, D, D, /*gstart=*/1 << 30);
}
